// Model_67087389163587
// MI455X (gfx1250) — matrix operand order repaired and checked by reading, not compiled, not run on hardware
//
#include <hip/hip_runtime.h>
#include <math.h>

// ---------------- model constants ----------------
#define D_MODEL  1024
#define NH       8
#define NKV      2
#define HD       128
#define INTER    8192
#define SEQ      1024
#define BATCH    2
#define NUM_CB   32
#define AUDIO_V  1024
#define MAXL     33
#define MLP_CH   2048
#define N_LAYERS 4

// ---------------- feature probes ----------------
#if defined(__has_builtin)
#  if __has_builtin(__builtin_amdgcn_global_load_async_to_lds_b128)
#    define HAVE_ASYNC_LDS 1
#  endif
#endif
#ifndef HAVE_ASYNC_LDS
#  define HAVE_ASYNC_LDS 0
#endif

// ---------------- types ----------------
typedef __attribute__((ext_vector_type(16))) __bf16         v16bf;
typedef __attribute__((ext_vector_type(8)))  float          v8f;
typedef __attribute__((ext_vector_type(16))) unsigned short v16us;
typedef __attribute__((ext_vector_type(8)))  unsigned short v8us;
typedef __attribute__((ext_vector_type(4)))  float          v4f;
typedef __attribute__((ext_vector_type(4)))  int            v4i;

#define WMMA_BF16(a, b, c) \
  __builtin_amdgcn_wmma_f32_16x16x32_bf16(false, (a), false, (b), (short)0, (c), false, false)

static __device__ __forceinline__ int imin(int a, int b) { return a < b ? a : b; }

static __device__ __forceinline__ unsigned short f2bf(float f) {
  union { float f; unsigned u; } c; c.f = f;
  unsigned r = c.u + 0x7FFFu + ((c.u >> 16) & 1u);   // round-to-nearest-even
  return (unsigned short)(r >> 16);
}
static __device__ __forceinline__ v16bf us2bf(v16us u) {
  union { v16us u; v16bf b; } c; c.u = u; return c.b;
}
// A fragment (16x32 bf16). lane<16: row=lane, k={0..15}; lane>=16: row=lane-16,
// k={16..31}.  caller passes row base and kb=(lane>>4)*8.
static __device__ __forceinline__ v16bf frag_a(const unsigned short* row, int kb) {
  v16us r;
#pragma unroll
  for (int i = 0; i < 8; ++i) { r[i] = row[2 * kb + i]; r[8 + i] = row[2 * kb + 8 + i]; }
  return us2bf(r);
}
// B fragment (32x16 bf16): lane holds column n=lane&15, k=(lane>>4)*16..+15, contiguous.
static __device__ __forceinline__ v16bf frag_b(const unsigned short* p) {
  v16us r;
#pragma unroll
  for (int i = 0; i < 16; ++i) r[i] = p[i];
  return us2bf(r);
}

#if HAVE_ASYNC_LDS
static __device__ __forceinline__ void async_b128(const void* g, void* l) {
  // builtin signature (per hipcc diagnostic): param 1 is v4i* (generic, non-const)
  __builtin_amdgcn_global_load_async_to_lds_b128((v4i*)g, (v4i*)l, 0, 0);
}
static __device__ __forceinline__ void wait_async0() {
#if __has_builtin(__builtin_amdgcn_s_wait_asynccnt)
  __builtin_amdgcn_s_wait_asynccnt(0);
#else
  asm volatile("s_wait_asynccnt 0x0" ::: "memory");
#endif
}
#endif

// ---------------- GEMM: C[M,N] (+)= A(bf16,MxK) * B(f32 weights,KxN) ----------------
#define GBK  32
#define GPAD 40   // LDS row stride in halves (80B: 16B aligned, spreads banks)

template <int ADDC>
__global__ __launch_bounds__(256)
void gemm_kernel(const unsigned short* __restrict__ A, int lda,
                 const float* __restrict__ Bw, int ldb,
                 float* __restrict__ C, int ldc,
                 int M, int N, int K)
{
  __shared__ unsigned short As[2][128 * GPAD];   // [row][k]
  __shared__ unsigned short Bs[2][128 * GPAD];   // transposed: [n][k]
  const int tid  = threadIdx.x;
  const int lane = tid & 31;
  const int wave = tid >> 5;
  const int wm   = (wave & 3) * 32;
  const int wn   = (wave >> 2) * 64;
  const int bm   = blockIdx.y * 128;
  const int bn   = blockIdx.x * 128;

  v8f acc[2][4];
#pragma unroll
  for (int i = 0; i < 2; ++i)
#pragma unroll
    for (int j = 0; j < 4; ++j)
#pragma unroll
      for (int r = 0; r < 8; ++r) acc[i][j][r] = 0.0f;

  const int ar  = tid >> 1;          // A stage: 2 threads per row, 16 halves each
  const int ac  = (tid & 1) * 16;
  const int bkr = tid >> 3;          // B stage: 8 threads per k-row, 16 floats each
  const int bcc = (tid & 7) * 16;
  int gr = bm + ar; if (gr >= M) gr = M - 1;     // clamp; OOB rows never stored
  const unsigned short* abase = A + (size_t)gr * lda + ac;
  const float* bbase = Bw + (size_t)bkr * ldb + bn + bcc;

  const int nt = K / GBK;
  v4f rb[4];
#if !HAVE_ASYNC_LDS
  v8us ra0, ra1;
#endif

  // ---- prologue: stage tile 0 ----
  {
#if HAVE_ASYNC_LDS
    async_b128(abase,     &As[0][ar * GPAD + ac]);
    async_b128(abase + 8, &As[0][ar * GPAD + ac + 8]);
#else
    v8us t0 = *(const v8us*)(abase);
    v8us t1 = *(const v8us*)(abase + 8);
    *(v8us*)(&As[0][ar * GPAD + ac])     = t0;
    *(v8us*)(&As[0][ar * GPAD + ac + 8]) = t1;
#endif
#pragma unroll
    for (int v = 0; v < 4; ++v) rb[v] = *(const v4f*)(bbase + v * 4);
#pragma unroll
    for (int v = 0; v < 4; ++v)
#pragma unroll
      for (int e = 0; e < 4; ++e)
        Bs[0][(bcc + v * 4 + e) * GPAD + bkr] = f2bf(rb[v][e]);
#if HAVE_ASYNC_LDS
    wait_async0();
#endif
  }
  __syncthreads();

  for (int kt = 0; kt < nt; ++kt) {
    const int cur = kt & 1;
    const int nxt = cur ^ 1;
    const bool more = (kt + 1) < nt;

    // kick off next-tile global traffic before the matrix work
    if (more) {
      const unsigned short* asrc = abase + (kt + 1) * GBK;
#if HAVE_ASYNC_LDS
      async_b128(asrc,     &As[nxt][ar * GPAD + ac]);
      async_b128(asrc + 8, &As[nxt][ar * GPAD + ac + 8]);
#else
      ra0 = *(const v8us*)(asrc);
      ra1 = *(const v8us*)(asrc + 8);
#endif
      const float* bsrc = bbase + (size_t)(kt + 1) * GBK * ldb;
#pragma unroll
      for (int v = 0; v < 4; ++v) rb[v] = *(const v4f*)(bsrc + v * 4);
      __builtin_prefetch(bsrc + (size_t)GBK * ldb, 0, 0);   // 2 tiles ahead
    }

    // compute on current buffers
    v16bf af[2];
#pragma unroll
    for (int i = 0; i < 2; ++i)
      af[i] = frag_a(&As[cur][(wm + i * 16 + (lane & 15)) * GPAD], (lane >> 4) * 8);
    v16bf bf[4];
#pragma unroll
    for (int j = 0; j < 4; ++j)
      bf[j] = frag_b(&Bs[cur][(wn + j * 16 + (lane & 15)) * GPAD + (lane >> 4) * 16]);
#pragma unroll
    for (int i = 0; i < 2; ++i)
#pragma unroll
      for (int j = 0; j < 4; ++j)
        acc[i][j] = WMMA_BF16(af[i], bf[j], acc[i][j]);

    // write next tile into the other buffer
    if (more) {
#if !HAVE_ASYNC_LDS
      *(v8us*)(&As[nxt][ar * GPAD + ac])     = ra0;
      *(v8us*)(&As[nxt][ar * GPAD + ac + 8]) = ra1;
#endif
#pragma unroll
      for (int v = 0; v < 4; ++v)
#pragma unroll
        for (int e = 0; e < 4; ++e)
          Bs[nxt][(bcc + v * 4 + e) * GPAD + bkr] = f2bf(rb[v][e]);
#if HAVE_ASYNC_LDS
      wait_async0();
#endif
    }
    __syncthreads();
  }

  // ---- epilogue (compile-time ADDC, fast path for full blocks) ----
  const bool mfull = (bm + 128) <= M;
#pragma unroll
  for (int i = 0; i < 2; ++i)
#pragma unroll
    for (int j = 0; j < 4; ++j) {
      const int m0 = bm + wm + i * 16 + (lane >> 4) * 8;
      const int n  = bn + wn + j * 16 + (lane & 15);
      float* p = C + (size_t)m0 * ldc + n;
      if (mfull) {
#pragma unroll
        for (int r = 0; r < 8; ++r) {
          float v = acc[i][j][r];
          if (ADDC) v += p[(size_t)r * ldc];
          p[(size_t)r * ldc] = v;
        }
      } else {
#pragma unroll
        for (int r = 0; r < 8; ++r) {
          if (m0 + r < M) {
            float v = acc[i][j][r];
            if (ADDC) v += p[(size_t)r * ldc];
            p[(size_t)r * ldc] = v;
          }
        }
      }
    }
}

// ---------------- flash attention with WMMA (causal) ----------------
#define KSP 136
#define VSP 40

__global__ __launch_bounds__(256)
void attn_kernel(const unsigned short* __restrict__ Qh,   // [B,NH,Sq,HD]
                 const unsigned short* __restrict__ Kh,   // [B,NKV,Skv,HD]
                 const unsigned short* __restrict__ Vh,   // [B,NKV,Skv,HD]
                 unsigned short* __restrict__ O,          // [B,Sq,NH,HD]
                 int Sq, int Skv)
{
  __shared__ unsigned short Ks[32 * KSP];        // [key][d]
  __shared__ unsigned short Vs[128 * VSP];       // transposed [d][key]
  __shared__ unsigned short Ps[8 * 16 * VSP];    // per-wave P slab

  const int bh  = blockIdx.y;
  const int b   = bh / NH, h = bh % NH;
  const int kvh = h / (NH / NKV);
  const int tid = threadIdx.x, lane = tid & 31, wave = tid >> 5;
  const int qrow0 = blockIdx.x * 128 + wave * 16;
  const float scale = 0.08838834764831845f;      // 1/sqrt(128)

  v16bf qf[4];
  {
    int qr = qrow0 + (lane & 15); if (qr >= Sq) qr = Sq - 1;
    const unsigned short* qp = Qh + ((size_t)(b * NH + h) * Sq + qr) * HD;
#pragma unroll
    for (int kk = 0; kk < 4; ++kk) qf[kk] = frag_a(qp + kk * 32, (lane >> 4) * 8);
  }

  float rmax[8], rsum[8];
  v8f oacc[8];
#pragma unroll
  for (int r = 0; r < 8; ++r) { rmax[r] = -INFINITY; rsum[r] = 0.f; }
#pragma unroll
  for (int f = 0; f < 8; ++f)
#pragma unroll
    for (int r = 0; r < 8; ++r) oacc[f][r] = 0.f;

  const int srow = tid >> 3;
  const int scol = (tid & 7) * 16;
  const int kend = imin(Skv, blockIdx.x * 128 + 128);   // causal upper bound

  for (int k0 = 0; k0 < kend; k0 += 32) {
    {
      int kidx = k0 + srow;
      if (kidx < Skv) {
        const unsigned short* kp = Kh + ((size_t)(b * NKV + kvh) * Skv + kidx) * HD + scol;
        const unsigned short* vp = Vh + ((size_t)(b * NKV + kvh) * Skv + kidx) * HD + scol;
#if HAVE_ASYNC_LDS
        async_b128(kp,     &Ks[srow * KSP + scol]);
        async_b128(kp + 8, &Ks[srow * KSP + scol + 8]);
#else
        v8us a0 = *(const v8us*)kp, a1 = *(const v8us*)(kp + 8);
        *(v8us*)(&Ks[srow * KSP + scol])     = a0;
        *(v8us*)(&Ks[srow * KSP + scol + 8]) = a1;
#endif
#pragma unroll
        for (int e = 0; e < 16; ++e) Vs[(scol + e) * VSP + srow] = vp[e];
      } else {
        v8us z = {0, 0, 0, 0, 0, 0, 0, 0};
        *(v8us*)(&Ks[srow * KSP + scol])     = z;
        *(v8us*)(&Ks[srow * KSP + scol + 8]) = z;
#pragma unroll
        for (int e = 0; e < 16; ++e) Vs[(scol + e) * VSP + srow] = 0;
      }
#if HAVE_ASYNC_LDS
      wait_async0();
#endif
    }
    __syncthreads();

    v8f sa[2];
#pragma unroll
    for (int j = 0; j < 2; ++j)
#pragma unroll
      for (int r = 0; r < 8; ++r) sa[j][r] = 0.f;
#pragma unroll
    for (int kk = 0; kk < 4; ++kk)
#pragma unroll
      for (int j = 0; j < 2; ++j) {
        v16bf bfrag = frag_b(&Ks[(j * 16 + (lane & 15)) * KSP + kk * 32 + (lane >> 4) * 16]);
        sa[j] = WMMA_BF16(qf[kk], bfrag, sa[j]);
      }

    float nm[8], corr[8];
#pragma unroll
    for (int r = 0; r < 8; ++r) {
      int qi = qrow0 + (lane >> 4) * 8 + r;
#pragma unroll
      for (int j = 0; j < 2; ++j) {
        int ki = k0 + j * 16 + (lane & 15);
        float sv = sa[j][r] * scale;
        if (ki > qi || ki >= Skv) sv = -INFINITY;
        sa[j][r] = sv;
      }
      float m = fmaxf(sa[0][r], sa[1][r]);
      m = fmaxf(m, __shfl_xor(m, 1, 16));
      m = fmaxf(m, __shfl_xor(m, 2, 16));
      m = fmaxf(m, __shfl_xor(m, 4, 16));
      m = fmaxf(m, __shfl_xor(m, 8, 16));
      nm[r] = fmaxf(rmax[r], m);
      float p0 = __expf(sa[0][r] - nm[r]);
      float p1 = __expf(sa[1][r] - nm[r]);
      sa[0][r] = p0; sa[1][r] = p1;
      float s = p0 + p1;
      s += __shfl_xor(s, 1, 16);
      s += __shfl_xor(s, 2, 16);
      s += __shfl_xor(s, 4, 16);
      s += __shfl_xor(s, 8, 16);
      corr[r] = __expf(rmax[r] - nm[r]);
      rsum[r] = rsum[r] * corr[r] + s;
      rmax[r] = nm[r];
    }
#pragma unroll
    for (int f = 0; f < 8; ++f)
#pragma unroll
      for (int r = 0; r < 8; ++r) oacc[f][r] *= corr[r];

    // route P through per-wave LDS slab: C-layout -> A-fragment layout
    unsigned short* pw = &Ps[wave * 16 * VSP];
#pragma unroll
    for (int j = 0; j < 2; ++j)
#pragma unroll
      for (int r = 0; r < 8; ++r)
        pw[((lane >> 4) * 8 + r) * VSP + j * 16 + (lane & 15)] = f2bf(sa[j][r]);
    v16bf pa = frag_a(&pw[(lane & 15) * VSP], (lane >> 4) * 8);

#pragma unroll
    for (int f = 0; f < 8; ++f) {
      v16bf vfrag = frag_b(&Vs[(f * 16 + (lane & 15)) * VSP + (lane >> 4) * 16]);
      oacc[f] = WMMA_BF16(pa, vfrag, oacc[f]);
    }
    __syncthreads();
  }

#pragma unroll
  for (int f = 0; f < 8; ++f)
#pragma unroll
    for (int r = 0; r < 8; ++r) {
      int qi = qrow0 + (lane >> 4) * 8 + r;
      if (qi < Sq) {
        int d = f * 16 + (lane & 15);
        O[(((size_t)b * Sq + qi) * NH + h) * HD + d] = f2bf(oacc[f][r] / rsum[r]);
      }
    }
}

// ---------------- rmsnorm -> bf16 ----------------
__global__ __launch_bounds__(256)
void rmsnorm_kernel(const float* __restrict__ x, const float* __restrict__ w,
                    unsigned short* __restrict__ out)
{
  const int row = blockIdx.x;
  const float* xr = x + (size_t)row * D_MODEL;
  __shared__ float red[8];
  const int lane = threadIdx.x & 31, wave = threadIdx.x >> 5;
  float ss = 0.f;
  for (int i = threadIdx.x; i < D_MODEL; i += 256) { float v = xr[i]; ss += v * v; }
  for (int off = 16; off > 0; off >>= 1) ss += __shfl_xor(ss, off, 32);
  if (lane == 0) red[wave] = ss;
  __syncthreads();
  if (threadIdx.x == 0) {
    float t = 0.f;
    for (int i = 0; i < 8; ++i) t += red[i];
    red[0] = rsqrtf(t / (float)D_MODEL + 1e-5f);
  }
  __syncthreads();
  float rs = red[0];
  for (int i = threadIdx.x; i < D_MODEL; i += 256)
    out[(size_t)row * D_MODEL + i] = f2bf(xr[i] * rs * w[i]);
}

// ---------------- embedding sum ----------------
__global__ __launch_bounds__(256)
void embed_kernel(const int* __restrict__ tokens, const float* __restrict__ mask,
                  const float* __restrict__ text_emb, const float* __restrict__ audio_emb,
                  float* __restrict__ h)
{
  const int bs = blockIdx.x;
  const int* tk = tokens + (size_t)bs * (NUM_CB + 1);
  const float* mk = mask + (size_t)bs * (NUM_CB + 1);
  float acc[4] = {0.f, 0.f, 0.f, 0.f};
  for (int c = 0; c < NUM_CB; ++c) {
    float m = mk[c];
    const float* e = audio_emb + ((size_t)tk[c] + (size_t)AUDIO_V * c) * D_MODEL;
#pragma unroll
    for (int j = 0; j < 4; ++j) acc[j] += m * e[threadIdx.x + 256 * j];
  }
  {
    float m = mk[NUM_CB];
    const float* e = text_emb + (size_t)tk[NUM_CB] * D_MODEL;
#pragma unroll
    for (int j = 0; j < 4; ++j) acc[j] += m * e[threadIdx.x + 256 * j];
  }
#pragma unroll
  for (int j = 0; j < 4; ++j) h[(size_t)bs * D_MODEL + threadIdx.x + 256 * j] = acc[j];
}

// ---------------- rope (llama3 scaling) + head-major bf16 pack ----------------
static __device__ __forceinline__ float rope_scaled_freq(int i) {
  float fr = __powf(500000.0f, -(2.0f * (float)i) / 128.0f);
  float wavelen = 6.28318530718f / fr;
  if (wavelen > 8192.0f) return fr / 32.0f;   // OLD_CTX / low_f
  if (wavelen < 2048.0f) return fr;           // OLD_CTX / high_f
  float smooth = (8192.0f / wavelen - 1.0f) / 3.0f;
  return (1.0f - smooth) * fr / 32.0f + smooth * fr;
}

__global__ __launch_bounds__(256)
void rope_pack_kernel(const float* __restrict__ qf, const float* __restrict__ kf,
                      const float* __restrict__ vf,
                      unsigned short* __restrict__ qh, unsigned short* __restrict__ kh,
                      unsigned short* __restrict__ vh, int S)
{
  const int bs = blockIdx.x;
  const int b = bs / S, s = bs % S;
  const float pos = (float)s;
  for (int p = threadIdx.x; p < NH * (HD / 2); p += 256) {
    int hh = p >> 6, i = p & 63;
    float fr = rope_scaled_freq(i), sn, cs;
    __sincosf(pos * fr, &sn, &cs);
    const float* src = qf + (size_t)bs * (NH * HD) + hh * HD;
    float x1 = src[2 * i], x2 = src[2 * i + 1];
    unsigned short* dst = qh + ((size_t)(b * NH + hh) * S + s) * HD;
    dst[2 * i]     = f2bf(x1 * cs - x2 * sn);
    dst[2 * i + 1] = f2bf(x2 * cs + x1 * sn);
  }
  for (int p = threadIdx.x; p < NKV * (HD / 2); p += 256) {
    int hh = p >> 6, i = p & 63;
    float fr = rope_scaled_freq(i), sn, cs;
    __sincosf(pos * fr, &sn, &cs);
    const float* src = kf + (size_t)bs * (NKV * HD) + hh * HD;
    float x1 = src[2 * i], x2 = src[2 * i + 1];
    unsigned short* dst = kh + ((size_t)(b * NKV + hh) * S + s) * HD;
    dst[2 * i]     = f2bf(x1 * cs - x2 * sn);
    dst[2 * i + 1] = f2bf(x2 * cs + x1 * sn);
  }
  for (int p = threadIdx.x; p < NKV * HD; p += 256) {
    int hh = p >> 7, d = p & 127;
    vh[((size_t)(b * NKV + hh) * S + s) * HD + d] =
        f2bf(vf[(size_t)bs * (NKV * HD) + hh * HD + d]);
  }
}

// ---------------- silu(g1)*g3 -> bf16 ----------------
__global__ __launch_bounds__(256)
void silumul_kernel(const float* __restrict__ g1, const float* __restrict__ g3,
                    unsigned short* __restrict__ out, size_t n)
{
  size_t i = (size_t)blockIdx.x * 256 + threadIdx.x;
  if (i < n) {
    float a = g1[i];
    out[i] = f2bf((a / (1.0f + __expf(-a))) * g3[i]);
  }
}

// ---------------- small helpers ----------------
__global__ __launch_bounds__(256)
void pack_rows_kernel(const float* __restrict__ seq, float* __restrict__ dst, int L)
{
  int b = blockIdx.x / L, l = blockIdx.x % L;
  const float* s = seq + ((size_t)b * MAXL + l) * D_MODEL;
  float* d = dst + (size_t)blockIdx.x * D_MODEL;
  for (int i = threadIdx.x; i < D_MODEL; i += 256) d[i] = s[i];
}

__global__ __launch_bounds__(256)
void gather_emb_kernel(const float* __restrict__ emb, const int* __restrict__ toks,
                       int tstride, int tcol, int off, unsigned short* __restrict__ dst)
{
  int b = blockIdx.x;
  int t = toks[b * tstride + tcol] + off;
  const float* e = emb + (size_t)t * D_MODEL;
  for (int i = threadIdx.x; i < D_MODEL; i += 256) dst[(size_t)b * D_MODEL + i] = f2bf(e[i]);
}

// ---------------- top-k sampling ----------------
static __device__ __forceinline__ float hash_uniform(unsigned fold, unsigned b, unsigned i) {
  unsigned long long x = 0x9E3779B97F4A7C15ull ^ ((unsigned long long)fold << 42)
                       ^ ((unsigned long long)b << 32) ^ (unsigned long long)i;
  x ^= x >> 30; x *= 0xBF58476D1CE4E5B9ull;
  x ^= x >> 27; x *= 0x94D049BB133111EBull;
  x ^= x >> 31;
  return ((float)(x >> 40) + 1.0f) * (1.0f / 16777216.0f);  // (0,1]
}

__global__ __launch_bounds__(256)
void sample_kernel(const float* __restrict__ logits, int rstride, int coff,
                   const float* __restrict__ temp_p, const int* __restrict__ topk_p,
                   int* __restrict__ out_tok, int ostride, int ocol, unsigned fold)
{
  const int b = blockIdx.x;
  const float* lg = logits + (size_t)b * rstride + coff;
  __shared__ float buf[AUDIO_V];
  __shared__ float keep[AUDIO_V];
  __shared__ float redf[8];
  __shared__ int   redi[8];
  const int tid = threadIdx.x, lane = tid & 31, wave = tid >> 5;
  float T = temp_p[0];
  int k = topk_p[0]; if (k > AUDIO_V) k = AUDIO_V; if (k < 1) k = 1;
  for (int i = tid; i < AUDIO_V; i += 256) { float v = lg[i] / T; buf[i] = v; keep[i] = v; }
  __syncthreads();

  float gmax = 0.f, kth = 0.f;
  for (int it = 0; it < k; ++it) {
    float m = -INFINITY; int mi = 0;
    for (int i = tid; i < AUDIO_V; i += 256)
      if (buf[i] > m) { m = buf[i]; mi = i; }
    for (int off = 16; off > 0; off >>= 1) {
      float om = __shfl_xor(m, off, 32);
      int   oi = __shfl_xor(mi, off, 32);
      if (om > m) { m = om; mi = oi; }
    }
    if (lane == 0) { redf[wave] = m; redi[wave] = mi; }
    __syncthreads();
    if (tid == 0) {
      float bm = redf[0]; int bi = redi[0];
      for (int wv = 1; wv < 8; ++wv) if (redf[wv] > bm) { bm = redf[wv]; bi = redi[wv]; }
      redf[0] = bm; buf[bi] = -INFINITY;
    }
    __syncthreads();
    float m2 = redf[0];
    if (it == 0) gmax = m2;
    kth = m2;
  }

  float bm = -INFINITY; int bi = 0;
  for (int i = tid; i < AUDIO_V; i += 256) {
    float l = keep[i];
    float p = (l < kth) ? 0.0f : __expf(l - gmax);
    float u = hash_uniform(fold, (unsigned)b, (unsigned)i);
    float sc = p / (-__logf(u) + 1e-20f);
    if (sc > bm) { bm = sc; bi = i; }
  }
  for (int off = 16; off > 0; off >>= 1) {
    float om = __shfl_xor(bm, off, 32);
    int   oi = __shfl_xor(bi, off, 32);
    if (om > bm) { bm = om; bi = oi; }
  }
  if (lane == 0) { redf[wave] = bm; redi[wave] = bi; }
  __syncthreads();
  if (tid == 0) {
    float fb = redf[0]; int fi = redi[0];
    for (int wv = 1; wv < 8; ++wv) if (redf[wv] > fb) { fb = redf[wv]; fi = redi[wv]; }
    out_tok[b * ostride + ocol] = fi;
  }
}

// ---------------- host orchestration ----------------
struct LayerW { const float *attn_norm,*mlp_norm,*w1,*w2,*w3,*wk,*wo,*wq,*wv; };
struct TfW { LayerW L[N_LAYERS]; const float* final_norm; };
struct Scratch {
  float *h, *qf, *kf, *vf, *g1, *g3, *seq, *dec_x;
  unsigned short *xn, *qh, *kh, *vh, *ob, *hb, *embA;
};

static inline void launch_gemm(hipStream_t st, const unsigned short* A, int lda,
                               const float* B, int ldb, float* C, int ldc,
                               int M, int N, int K, int addC) {
  dim3 grid(N / 128, (M + 127) / 128);
  if (addC)
    gemm_kernel<1><<<grid, 256, 0, st>>>(A, lda, B, ldb, C, ldc, M, N, K);
  else
    gemm_kernel<0><<<grid, 256, 0, st>>>(A, lda, B, ldb, C, ldc, M, N, K);
}

static void run_tf(hipStream_t st, const TfW& P, float* x, int Bb, int S, const Scratch& sc)
{
  const int M = Bb * S;
  for (int l = 0; l < N_LAYERS; ++l) {
    const LayerW& W = P.L[l];
    rmsnorm_kernel<<<M, 256, 0, st>>>(x, W.attn_norm, sc.xn);
    launch_gemm(st, sc.xn, D_MODEL, W.wq, D_MODEL, sc.qf, D_MODEL, M, D_MODEL, D_MODEL, 0);
    launch_gemm(st, sc.xn, D_MODEL, W.wk, NKV * HD, sc.kf, NKV * HD, M, NKV * HD, D_MODEL, 0);
    launch_gemm(st, sc.xn, D_MODEL, W.wv, NKV * HD, sc.vf, NKV * HD, M, NKV * HD, D_MODEL, 0);
    rope_pack_kernel<<<M, 256, 0, st>>>(sc.qf, sc.kf, sc.vf, sc.qh, sc.kh, sc.vh, S);
    attn_kernel<<<dim3((S + 127) / 128, Bb * NH), 256, 0, st>>>(sc.qh, sc.kh, sc.vh, sc.ob, S, S);
    launch_gemm(st, sc.ob, D_MODEL, W.wo, D_MODEL, x, D_MODEL, M, D_MODEL, D_MODEL, 1);
    rmsnorm_kernel<<<M, 256, 0, st>>>(x, W.mlp_norm, sc.xn);
    for (int c = 0; c < INTER; c += MLP_CH) {
      launch_gemm(st, sc.xn, D_MODEL, W.w1 + c, INTER, sc.g1, MLP_CH, M, MLP_CH, D_MODEL, 0);
      launch_gemm(st, sc.xn, D_MODEL, W.w3 + c, INTER, sc.g3, MLP_CH, M, MLP_CH, D_MODEL, 0);
      silumul_kernel<<<(M * MLP_CH) / 256, 256, 0, st>>>(sc.g1, sc.g3, sc.hb, (size_t)M * MLP_CH);
      launch_gemm(st, sc.hb, MLP_CH, W.w2 + (size_t)c * D_MODEL, D_MODEL, x, D_MODEL,
                  M, D_MODEL, MLP_CH, 1);
    }
  }
}

extern "C" void kernel_launch(void* const* d_in, const int* in_sizes, int n_in,
                              void* d_out, int out_size, void* d_ws, size_t ws_size,
                              hipStream_t stream)
{
  (void)in_sizes; (void)n_in; (void)out_size; (void)ws_size;
  const int*   tokens      = (const int*)  d_in[0];
  const float* tokens_mask = (const float*)d_in[1];
  // d_in[2] = input_pos (arange -> causal by index)
  const float* temperature = (const float*)d_in[3];
  const int*   topk        = (const int*)  d_in[4];
  const float* text_emb    = (const float*)d_in[5];
  const float* audio_emb   = (const float*)d_in[6];
  const float* proj_w      = (const float*)d_in[7];
  const float* cb0_w       = (const float*)d_in[8];
  const float* audio_head  = (const float*)d_in[9];

  // params: jax pytree order (sorted dict keys): final_norm, then per layer
  // attn_norm, mlp_norm, w1, w2, w3, wk, wo, wq, wv
  TfW bb, dec;
  TfW* tfs[2] = {&bb, &dec};
  int idx = 10;
  for (int ti = 0; ti < 2; ++ti) {
    TfW* t = tfs[ti];
    t->final_norm = (const float*)d_in[idx++];
    for (int l = 0; l < N_LAYERS; ++l) {
      LayerW& L = t->L[l];
      L.attn_norm = (const float*)d_in[idx++];
      L.mlp_norm  = (const float*)d_in[idx++];
      L.w1 = (const float*)d_in[idx++];
      L.w2 = (const float*)d_in[idx++];
      L.w3 = (const float*)d_in[idx++];
      L.wk = (const float*)d_in[idx++];
      L.wo = (const float*)d_in[idx++];
      L.wq = (const float*)d_in[idx++];
      L.wv = (const float*)d_in[idx++];
    }
  }

  // workspace carve
  char* wsp = (char*)d_ws;
  auto carve = [&](size_t bytes) { char* p = wsp; wsp += (bytes + 255) & ~(size_t)255; return p; };
  Scratch sc;
  const size_t Mmax = (size_t)BATCH * SEQ;
  sc.h     = (float*)carve(Mmax * D_MODEL * 4);
  sc.qf    = (float*)carve(Mmax * D_MODEL * 4);
  sc.kf    = (float*)carve(Mmax * NKV * HD * 4);
  sc.vf    = (float*)carve(Mmax * NKV * HD * 4);
  sc.g1    = (float*)carve(Mmax * MLP_CH * 4);
  sc.g3    = (float*)carve(Mmax * MLP_CH * 4);
  sc.seq   = (float*)carve((size_t)BATCH * MAXL * D_MODEL * 4);
  sc.dec_x = (float*)carve((size_t)BATCH * MAXL * D_MODEL * 4);
  sc.xn    = (unsigned short*)carve(Mmax * D_MODEL * 2);
  sc.qh    = (unsigned short*)carve(Mmax * D_MODEL * 2);
  sc.kh    = (unsigned short*)carve(Mmax * NKV * HD * 2);
  sc.vh    = (unsigned short*)carve(Mmax * NKV * HD * 2);
  sc.ob    = (unsigned short*)carve(Mmax * D_MODEL * 2);
  sc.hb    = (unsigned short*)carve(Mmax * MLP_CH * 2);
  sc.embA  = (unsigned short*)carve((size_t)BATCH * D_MODEL * 2);

  int*   out_tok    = (int*)d_out;                           // (B, 32) int32
  float* out_logits = (float*)d_out + BATCH * NUM_CB;        // (B, 32, 1024) f32

  // 1) embedding sum
  embed_kernel<<<BATCH * SEQ, 256, 0, stream>>>(tokens, tokens_mask, text_emb, audio_emb, sc.h);
  // 2) backbone transformer + final norm (bf16 normalized activations in sc.xn)
  run_tf(stream, bb, sc.h, BATCH, SEQ, sc);
  rmsnorm_kernel<<<BATCH * SEQ, 256, 0, stream>>>(sc.h, bb.final_norm, sc.xn);

  // 3) codebook 0: last row per batch = xn rows (b*SEQ + SEQ-1), lda = SEQ*D
  const unsigned short* lastRows = sc.xn + (size_t)(SEQ - 1) * D_MODEL;
  launch_gemm(stream, lastRows, SEQ * D_MODEL, cb0_w, AUDIO_V,
              out_logits, NUM_CB * AUDIO_V, BATCH, AUDIO_V, D_MODEL, 0);
  sample_kernel<<<BATCH, 256, 0, stream>>>(out_logits, NUM_CB * AUDIO_V, 0,
                                           temperature, topk, out_tok, NUM_CB, 0, 0u);

  // 4) seq = [last_h, audio_emb[c0]] @ projection_w
  launch_gemm(stream, lastRows, SEQ * D_MODEL, proj_w, D_MODEL,
              sc.seq, MAXL * D_MODEL, BATCH, D_MODEL, D_MODEL, 0);
  gather_emb_kernel<<<BATCH, 256, 0, stream>>>(audio_emb, out_tok, NUM_CB, 0, 0, sc.embA);
  launch_gemm(stream, sc.embA, D_MODEL, proj_w, D_MODEL,
              sc.seq + D_MODEL, MAXL * D_MODEL, BATCH, D_MODEL, D_MODEL, 0);

  // 5) decoder loop over codebooks 1..31
  int L = 2;
  for (int i = 1; i < NUM_CB; ++i) {
    pack_rows_kernel<<<BATCH * L, 256, 0, stream>>>(sc.seq, sc.dec_x, L);
    run_tf(stream, dec, sc.dec_x, BATCH, L, sc);
    rmsnorm_kernel<<<BATCH * L, 256, 0, stream>>>(sc.dec_x, dec.final_norm, sc.xn);
    launch_gemm(stream, sc.xn + (size_t)(L - 1) * D_MODEL, L * D_MODEL,
                audio_head + (size_t)(i - 1) * D_MODEL * AUDIO_V, AUDIO_V,
                out_logits + (size_t)i * AUDIO_V, NUM_CB * AUDIO_V,
                BATCH, AUDIO_V, D_MODEL, 0);
    sample_kernel<<<BATCH, 256, 0, stream>>>(out_logits, NUM_CB * AUDIO_V, i * AUDIO_V,
                                             temperature, topk, out_tok, NUM_CB, i, (unsigned)i);
    gather_emb_kernel<<<BATCH, 256, 0, stream>>>(audio_emb, out_tok, NUM_CB, i,
                                                 i * AUDIO_V, sc.embA);
    launch_gemm(stream, sc.embA, D_MODEL, proj_w, D_MODEL,
                sc.seq + (size_t)L * D_MODEL, MAXL * D_MODEL, BATCH, D_MODEL, D_MODEL, 0);
    ++L;
  }
}
